// DifferentPooling_60249801228692
// MI455X (gfx1250) — compile-verified, matrix operand order checked by reading, not run on hardware
//
#include <hip/hip_runtime.h>

#define HID 128

typedef __attribute__((ext_vector_type(16))) __bf16 v16bf;
typedef __attribute__((ext_vector_type(8)))  float  v8f;

union Frag16 { unsigned int u32[8]; v16bf v; };

__device__ __forceinline__ unsigned short f2bf(float f) {
  unsigned int u = __float_as_uint(f);
  u += 0x7FFFu + ((u >> 16) & 1u);           // round-to-nearest-even
  return (unsigned short)(u >> 16);
}

// monotone float max/min via integer atomics (lowers to global_atomic_max/min)
__device__ __forceinline__ void atomicMaxF(float* addr, float val) {
  if (val >= 0.f) atomicMax((int*)addr, __float_as_int(val));
  else            atomicMin((unsigned int*)addr, __float_as_uint(val));
}

// gfx1250 async copy: global -> LDS, 16 bytes per lane, tracked by ASYNCcnt.
__device__ __forceinline__ void async_copy_b128(void* lds_ptr, const void* gptr) {
  unsigned lds_off = (unsigned)(unsigned long long)lds_ptr;  // LDS aperture: offset in addr[31:0]
  asm volatile("global_load_async_to_lds_b128 %0, %1, off"
               :: "v"(lds_off), "v"(gptr) : "memory");
}
__device__ __forceinline__ void async_wait0() {
  asm volatile("s_wait_asynccnt 0" ::: "memory");
}

// ---------------------------------------------------------------- utilities
__global__ void k_fill(float* __restrict__ p, float v, int n) {
  int i = blockIdx.x * blockDim.x + threadIdx.x;
  if (i < n) p[i] = v;
}

__global__ void k_degree(const int* __restrict__ src, const int* __restrict__ dst,
                         float* __restrict__ ds_, float* __restrict__ dd_, int E_) {
  int i = blockIdx.x * blockDim.x + threadIdx.x;
  if (i < E_) {
    atomicAdd(&ds_[src[i]], 1.f);
    atomicAdd(&dd_[dst[i]], 1.f);
  }
}

__global__ void k_invsqrt(float* __restrict__ p, int n) {
  int i = blockIdx.x * blockDim.x + threadIdx.x;
  if (i < n) p[i] = rsqrtf(fmaxf(p[i], 1.f));
}

// W[K][Nc] fp32  ->  Wt[Nc][K] bf16 (transposed, so B fragments read contiguous K)
__global__ void k_w2bf_t(const float* __restrict__ W, unsigned short* __restrict__ Wt,
                         int K, int Nc) {
  int i = blockIdx.x * blockDim.x + threadIdx.x;
  if (i < K * Nc) {
    int k = i / Nc, n = i % Nc;
    Wt[n * K + k] = f2bf(W[i]);
  }
}

// ---------------------------------------------------------------- WMMA GEMM
// Y[nrows][128] = act( (X * rowscale) @ W + bias ),  W given transposed bf16.
// Block: 256 threads = 8 wave32, computes a 128x128 output tile.
__global__ __launch_bounds__(256) void gemm128_wmma(
    const float* __restrict__ X, const float* __restrict__ rowscale,
    const unsigned int* __restrict__ Wt,   // [128][64] dwords = bf16 pairs along K
    const float* __restrict__ bias, float* __restrict__ Y,
    int nrows, int relu) {
  __shared__ unsigned int xt[128 * 64];   // X tile, bf16 pairs: row-major, 64 dwords/row
  __shared__ unsigned int wt[128 * 64];   // W^T tile (row = out col n)

  const int t = threadIdx.x;
  const int rowBase = blockIdx.x * 128;

  // Stage W via async global->LDS B128 copies (no VGPR round-trip).
#pragma unroll
  for (int j = 0; j < 8; ++j) {
    int o = (t + j * 256) * 16;           // byte offset, 32KB total
    async_copy_b128((char*)wt + o, (const char*)Wt + o);
  }

  // Stage X with fp32->bf16 conversion (VALU work, so manual path).
  for (int i = t; i < 128 * 64; i += 256) {
    int r = i >> 6, cp = i & 63;
    int row = rowBase + r;
    float a = 0.f, b = 0.f;
    if (row < nrows) {
      const float2 v = *(const float2*)(X + (size_t)row * HID + cp * 2);
      float s = rowscale ? rowscale[row] : 1.0f;
      a = v.x * s; b = v.y * s;
    }
    xt[i] = (unsigned int)f2bf(a) | ((unsigned int)f2bf(b) << 16);
  }
  async_wait0();        // this wave's async copies done
  __syncthreads();      // all waves' copies + ds stores visible

  const int wid   = t >> 5;
  const int lane  = t & 31;
  const int lhalf = lane >> 4;   // ISA 7.12.2: lanes 16-31 hold K+8 within each 16-chunk
  const int l16   = lane & 15;
  const int arow  = wid * 16 + l16;
  const bool full = (rowBase + 128) <= nrows;   // uniform: all 391-1 blocks take fast path

  // Preload all 4 K-block A fragments; reuse across the 8 N tiles.
  Frag16 afr[4];
#pragma unroll
  for (int kb = 0; kb < 4; ++kb)
#pragma unroll
    for (int v = 0; v < 8; ++v) {
      int kw = 16 * kb + lhalf * 4 + ((v < 4) ? v : 4 + v);
      afr[kb].u32[v] = xt[arow * 64 + kw];
    }

#pragma unroll
  for (int nt = 0; nt < 8; ++nt) {
    v8f acc = {0.f, 0.f, 0.f, 0.f, 0.f, 0.f, 0.f, 0.f};
    const int bcol = nt * 16 + l16;
#pragma unroll
    for (int kb = 0; kb < 4; ++kb) {
      Frag16 bfr;
#pragma unroll
      for (int v = 0; v < 8; ++v) {
        int kw = 16 * kb + lhalf * 4 + ((v < 4) ? v : 4 + v);
        bfr.u32[v] = wt[bcol * 64 + kw];
      }
      acc = __builtin_amdgcn_wmma_f32_16x16x32_bf16(
          false, afr[kb].v, false, bfr.v, (short)0, acc, false, false);
    }
    const int col = nt * 16 + l16;
    const float bv = bias ? bias[col] : 0.f;
    float* yp = Y + (size_t)(rowBase + wid * 16 + 8 * lhalf) * HID + col;
    if (full) {
#pragma unroll
      for (int r = 0; r < 8; ++r) {       // unconditional stores (hot path)
        float v = acc[r] + bv;
        if (relu) v = v > 0.f ? v : 0.f;
        yp[(size_t)r * HID] = v;
      }
    } else {
#pragma unroll
      for (int r = 0; r < 8; ++r) {       // remainder block only
        int row = rowBase + wid * 16 + r + 8 * lhalf;   // D layout (ISA 7.12.2)
        if (row < nrows) {
          float v = acc[r] + bv;
          if (relu) v = v > 0.f ? v : 0.f;
          Y[(size_t)row * HID + col] = v;
        }
      }
    }
  }
}

// ---------------------------------------------------------------- edge ops
// agg[dst] += m[src]  (one wave per edge, float4 lanes)
__global__ __launch_bounds__(256) void k_scatter_add(
    const float* __restrict__ M_, const int* __restrict__ src,
    const int* __restrict__ dst, float* __restrict__ agg, int E_) {
  int e = blockIdx.x * 8 + (threadIdx.x >> 5);
  if (e >= E_) return;
  int lane = threadIdx.x & 31;
  int s = src[e], d = dst[e];
  const float4 v = *(const float4*)(M_ + (size_t)s * HID + lane * 4);
  float* o = agg + (size_t)d * HID + lane * 4;
  atomicAdd(o + 0, v.x); atomicAdd(o + 1, v.y);
  atomicAdd(o + 2, v.z); atomicAdd(o + 3, v.w);
}

__global__ void k_gc_finish(const float* __restrict__ agg, const float* __restrict__ nd,
                            const float* __restrict__ bias, float* __restrict__ h, int Nn) {
  int idx = blockIdx.x * blockDim.x + threadIdx.x;
  if (idx >= Nn * 32) return;
  int row = idx >> 5, c4 = idx & 31;
  float s = nd[row];
  float4 v = *(const float4*)(agg + (size_t)row * HID + c4 * 4);
  const float4 b = *(const float4*)(bias + c4 * 4);
  v.x = fmaxf(v.x * s + b.x, 0.f); v.y = fmaxf(v.y * s + b.y, 0.f);
  v.z = fmaxf(v.z * s + b.z, 0.f); v.w = fmaxf(v.w * s + b.w, 0.f);
  *(float4*)(h + (size_t)row * HID + c4 * 4) = v;
}

// logit[e][h] = sum_d( leaky_relu(fs[src][h,d]+fd[dst][h,d]) * attn[h,d] )
__global__ void k_edge_logit(const float* __restrict__ fs, const float* __restrict__ fd,
                             const int* __restrict__ src, const int* __restrict__ dst,
                             const float* __restrict__ attn, float* __restrict__ elog, int E_) {
  int idx = blockIdx.x * blockDim.x + threadIdx.x;
  if (idx >= E_ * 8) return;
  int e = idx >> 3, h = idx & 7;
  const float4* a4 = (const float4*)(fs + (size_t)src[e] * HID + h * 16);
  const float4* b4 = (const float4*)(fd + (size_t)dst[e] * HID + h * 16);
  const float4* w4 = (const float4*)(attn + h * 16);
  float acc = 0.f;
#pragma unroll
  for (int j = 0; j < 4; ++j) {
    float4 a = a4[j], b = b4[j], w = w4[j];
    float x;
    x = a.x + b.x; x = x > 0.f ? x : 0.2f * x; acc += x * w.x;
    x = a.y + b.y; x = x > 0.f ? x : 0.2f * x; acc += x * w.y;
    x = a.z + b.z; x = x > 0.f ? x : 0.2f * x; acc += x * w.z;
    x = a.w + b.w; x = x > 0.f ? x : 0.2f * x; acc += x * w.w;
  }
  elog[idx] = acc;
}

__global__ void k_edge_max(const float* __restrict__ elog, const int* __restrict__ dst,
                           float* __restrict__ emax, int E_) {
  int idx = blockIdx.x * blockDim.x + threadIdx.x;
  if (idx >= E_ * 8) return;
  int e = idx >> 3, h = idx & 7;
  atomicMaxF(&emax[(size_t)dst[e] * 8 + h], elog[idx]);
}

// p = exp(logit - max[dst]); sum[dst] += p; overwrite elog with p
__global__ void k_edge_expsum(float* __restrict__ elog, const int* __restrict__ dst,
                              const float* __restrict__ emax, float* __restrict__ esum, int E_) {
  int idx = blockIdx.x * blockDim.x + threadIdx.x;
  if (idx >= E_ * 8) return;
  int e = idx >> 3, h = idx & 7;
  int d = dst[e];
  float p = __expf(elog[idx] - emax[(size_t)d * 8 + h]);
  elog[idx] = p;
  atomicAdd(&esum[(size_t)d * 8 + h], p);
}

// out[dst] += (p/sum[dst]) * fs[src]   (one wave per edge; head = lane/4)
__global__ __launch_bounds__(256) void k_wscatter(
    const float* __restrict__ fs, const int* __restrict__ src, const int* __restrict__ dst,
    const float* __restrict__ p, const float* __restrict__ esum,
    float* __restrict__ out, int E_) {
  int e = blockIdx.x * 8 + (threadIdx.x >> 5);
  if (e >= E_) return;
  int lane = threadIdx.x & 31;
  int s = src[e], d = dst[e];
  int h = lane >> 2;
  float alpha = p[(size_t)e * 8 + h] / esum[(size_t)d * 8 + h];
  const float4 v = *(const float4*)(fs + (size_t)s * HID + lane * 4);
  float* o = out + (size_t)d * HID + lane * 4;
  atomicAdd(o + 0, alpha * v.x); atomicAdd(o + 1, alpha * v.y);
  atomicAdd(o + 2, alpha * v.z); atomicAdd(o + 3, alpha * v.w);
}

__global__ void k_gat_finish(const float* __restrict__ acc, const float* __restrict__ hin,
                             float* __restrict__ hout, int Nn) {
  int idx = blockIdx.x * blockDim.x + threadIdx.x;
  if (idx >= Nn * 32) return;
  size_t o = (size_t)(idx >> 5) * HID + (idx & 31) * 4;
  float4 a = *(const float4*)(acc + o);
  float4 b = *(const float4*)(hin + o);
  a.x = fmaxf(a.x + b.x, 0.f); a.y = fmaxf(a.y + b.y, 0.f);
  a.z = fmaxf(a.z + b.z, 0.f); a.w = fmaxf(a.w + b.w, 0.f);
  *(float4*)(hout + o) = a;
}

__global__ void k_pool(const float* __restrict__ h, const int* __restrict__ n2g,
                       float* __restrict__ hg, int Nn) {
  int idx = blockIdx.x * blockDim.x + threadIdx.x;
  if (idx >= Nn * 32) return;
  int row = idx >> 5, c4 = idx & 31;
  int g = n2g[row];
  const float4 v = *(const float4*)(h + (size_t)row * HID + c4 * 4);
  float* o = hg + (size_t)g * HID + c4 * 4;
  atomicMaxF(o + 0, v.x); atomicMaxF(o + 1, v.y);
  atomicMaxF(o + 2, v.z); atomicMaxF(o + 3, v.w);
}

// small tail layers: C[M][Nc] = act(A[M][K] @ B[K][Nc] + bias)
__global__ void k_naive_gemm(const float* __restrict__ A, const float* __restrict__ B,
                             const float* __restrict__ bias, float* __restrict__ C,
                             int M, int K, int Nc, int relu) {
  int idx = blockIdx.x * blockDim.x + threadIdx.x;
  if (idx >= M * Nc) return;
  int r = idx / Nc, c = idx % Nc;
  float acc = bias ? bias[c] : 0.f;
  for (int k = 0; k < K; ++k) acc += A[r * K + k] * B[k * Nc + c];
  if (relu) acc = fmaxf(acc, 0.f);
  C[idx] = acc;
}

// ---------------------------------------------------------------- launcher
extern "C" void kernel_launch(void* const* d_in, const int* in_sizes, int n_in,
                              void* d_out, int out_size, void* d_ws, size_t ws_size,
                              hipStream_t stream) {
  const float* feature = (const float*)d_in[0];
  const float* W_gc1 = (const float*)d_in[1];
  const float* b_gc1 = (const float*)d_in[2];
  const float* W_gc2 = (const float*)d_in[3];
  const float* b_gc2 = (const float*)d_in[4];
  const float* W_src = (const float*)d_in[5];
  const float* b_src = (const float*)d_in[6];
  const float* W_dst = (const float*)d_in[7];
  const float* b_dst = (const float*)d_in[8];
  const float* attn  = (const float*)d_in[9];
  const float* Wc1 = (const float*)d_in[10];
  const float* bc1 = (const float*)d_in[11];
  const float* Wc2 = (const float*)d_in[12];
  const float* bc2 = (const float*)d_in[13];
  const float* Wc3 = (const float*)d_in[14];
  const float* bc3 = (const float*)d_in[15];
  const int* src = (const int*)d_in[16];
  const int* dst = (const int*)d_in[17];
  const int* n2g = (const int*)d_in[18];

  const int N = in_sizes[0] / HID;
  const int E = in_sizes[16];
  const int G = out_size / 256;
  const int TB = 256;
  auto cdiv = [](int a, int b) { return (a + b - 1) / b; };

  // workspace carve (256B aligned)
  char* base = (char*)d_ws;
  size_t off = 0;
  auto carve = [&](size_t bytes) -> void* {
    void* p = base + off;
    off += (bytes + 255) & ~(size_t)255;
    return p;
  };
  float* ns = (float*)carve((size_t)N * 4);
  float* nd = (float*)carve((size_t)N * 4);
  float* hA = (float*)carve((size_t)N * HID * 4);
  float* hB = (float*)carve((size_t)N * HID * 4);
  float* t1 = (float*)carve((size_t)N * HID * 4);   // m / fs
  float* t2 = (float*)carve((size_t)N * HID * 4);   // agg / fd
  float* t3 = (float*)carve((size_t)N * HID * 4);   // GAT accum
  float* elog = (float*)carve((size_t)E * 8 * 4);
  float* emax = (float*)carve((size_t)N * 8 * 4);
  float* esum = (float*)carve((size_t)N * 8 * 4);
  unsigned short* wtA = (unsigned short*)carve((size_t)HID * HID * 2);
  unsigned short* wtB = (unsigned short*)carve((size_t)HID * HID * 2);
  float* hg = (float*)carve((size_t)G * HID * 4);
  float* z1 = (float*)carve((size_t)G * HID * 4);
  float* z2 = (float*)carve((size_t)G * 64 * 4);
  (void)ws_size; (void)n_in;

  // ---- degree norms
  k_fill<<<cdiv(N, TB), TB, 0, stream>>>(ns, 0.f, N);
  k_fill<<<cdiv(N, TB), TB, 0, stream>>>(nd, 0.f, N);
  k_degree<<<cdiv(E, TB), TB, 0, stream>>>(src, dst, ns, nd, E);
  k_invsqrt<<<cdiv(N, TB), TB, 0, stream>>>(ns, N);
  k_invsqrt<<<cdiv(N, TB), TB, 0, stream>>>(nd, N);

  // ---- GraphConv x2
  auto graph_conv = [&](const float* hin, const float* Wg, const float* bg, float* hout) {
    k_w2bf_t<<<cdiv(HID * HID, TB), TB, 0, stream>>>(Wg, wtA, HID, HID);
    gemm128_wmma<<<cdiv(N, 128), TB, 0, stream>>>(hin, ns, (const unsigned int*)wtA,
                                                  nullptr, t1, N, 0);
    k_fill<<<cdiv(N * HID, TB), TB, 0, stream>>>(t2, 0.f, N * HID);
    k_scatter_add<<<cdiv(E, 8), TB, 0, stream>>>(t1, src, dst, t2, E);
    k_gc_finish<<<cdiv(N * 32, TB), TB, 0, stream>>>(t2, nd, bg, hout, N);
  };
  graph_conv(feature, W_gc1, b_gc1, hA);
  graph_conv(hA, W_gc2, b_gc2, hB);

  // ---- GATv2 x3
  auto gat = [&](const float* hin, int layer, float* hout) {
    k_w2bf_t<<<cdiv(HID * HID, TB), TB, 0, stream>>>(W_src + (size_t)layer * HID * HID, wtA, HID, HID);
    k_w2bf_t<<<cdiv(HID * HID, TB), TB, 0, stream>>>(W_dst + (size_t)layer * HID * HID, wtB, HID, HID);
    gemm128_wmma<<<cdiv(N, 128), TB, 0, stream>>>(hin, nullptr, (const unsigned int*)wtA,
                                                  b_src + layer * HID, t1, N, 0);
    gemm128_wmma<<<cdiv(N, 128), TB, 0, stream>>>(hin, nullptr, (const unsigned int*)wtB,
                                                  b_dst + layer * HID, t2, N, 0);
    k_edge_logit<<<cdiv(E * 8, TB), TB, 0, stream>>>(t1, t2, src, dst,
                                                     attn + layer * HID, elog, E);
    k_fill<<<cdiv(N * 8, TB), TB, 0, stream>>>(emax, -3.0e38f, N * 8);
    k_edge_max<<<cdiv(E * 8, TB), TB, 0, stream>>>(elog, dst, emax, E);
    k_fill<<<cdiv(N * 8, TB), TB, 0, stream>>>(esum, 0.f, N * 8);
    k_edge_expsum<<<cdiv(E * 8, TB), TB, 0, stream>>>(elog, dst, emax, esum, E);
    k_fill<<<cdiv(N * HID, TB), TB, 0, stream>>>(t3, 0.f, N * HID);
    k_wscatter<<<cdiv(E, 8), TB, 0, stream>>>(t1, src, dst, elog, esum, t3, E);
    k_gat_finish<<<cdiv(N * 32, TB), TB, 0, stream>>>(t3, hin, hout, N);
  };
  gat(hB, 0, hA);
  gat(hA, 1, hB);
  gat(hB, 2, hA);

  // ---- graph max-pool
  k_fill<<<cdiv(G * HID, TB), TB, 0, stream>>>(hg, -3.0e38f, G * HID);
  k_pool<<<cdiv(N * 32, TB), TB, 0, stream>>>(hA, n2g, hg, N);

  // ---- classifier MLP
  k_w2bf_t<<<cdiv(HID * HID, TB), TB, 0, stream>>>(Wc1, wtA, HID, HID);
  gemm128_wmma<<<cdiv(G, 128), TB, 0, stream>>>(hg, nullptr, (const unsigned int*)wtA,
                                                bc1, z1, G, 1);
  k_naive_gemm<<<cdiv(G * 64, TB), TB, 0, stream>>>(z1, Wc2, bc2, z2, G, HID, 64, 1);
  k_naive_gemm<<<cdiv(G * 256, TB), TB, 0, stream>>>(z2, Wc3, bc3, (float*)d_out,
                                                     G, 64, 256, 0);
}
